// DTM_16423954940063
// MI455X (gfx1250) — matrix operand order repaired and checked by reading, not compiled, not run on hardware
//
#include <hip/hip_runtime.h>
#include <hip/hip_bf16.h>

typedef __attribute__((ext_vector_type(16))) _Float16 v16h;
typedef __attribute__((ext_vector_type(8)))  _Float16 v8h;
typedef __attribute__((ext_vector_type(8)))  float    v8f;

// Problem constants
#define BB   16
#define CIN  64
#define HH   64
#define WW   64
#define COUT 128
#define KTAP 9
#define KDIM 576            // CIN * KTAP, GEMM reduction
#define KSTEPS 18           // KDIM / 32
#define LDSROW 584          // 576 + 8 pad (f16 elements) per A-tile row

// ---------------------------------------------------------------------------
// Pack f32 weights into WMMA B-fragment order (f16):
//   Bbuf[((nt*18 + ks)*32 + lane)*16 + j] = W[k][n]
//   n = nt*16 + lane%16,  k = ks*32 + (lane/16)*16 + j
//   K ordering: k = tap*64 + ci  (tap-major, channel-minor)
// ---------------------------------------------------------------------------
__global__ void pack_weights_kernel(const float* __restrict__ w_p,
                                    const float* __restrict__ w_m,
                                    const float* __restrict__ w_d,
                                    _Float16* __restrict__ Bpm,
                                    _Float16* __restrict__ Bd) {
    int i0 = blockIdx.x * blockDim.x + threadIdx.x;
    int stride = gridDim.x * blockDim.x;
    // Bd: 8 ntiles * 18 ksteps * 32 lanes * 16 = 73728 f16
    for (int idx = i0; idx < 8 * KSTEPS * 32 * 16; idx += stride) {
        int j = idx & 15;
        int lane = (idx >> 4) & 31;
        int ks = (idx >> 9) % KSTEPS;
        int nt = idx / (16 * 32 * KSTEPS);
        int n = nt * 16 + (lane & 15);
        int k = ks * 32 + ((lane >> 4) << 4) + j;
        int ci = k & 63, t = k >> 6;
        Bd[idx] = (_Float16)w_d[n * KDIM + ci * 9 + t];
    }
    // Bpm: 2 ntiles (channels 0..17 offset, 18..26 mask, 27..31 zero)
    for (int idx = i0; idx < 2 * KSTEPS * 32 * 16; idx += stride) {
        int j = idx & 15;
        int lane = (idx >> 4) & 31;
        int ks = (idx >> 9) % KSTEPS;
        int nt = idx / (16 * 32 * KSTEPS);
        int n = nt * 16 + (lane & 15);
        int k = ks * 32 + ((lane >> 4) << 4) + j;
        int ci = k & 63, t = k >> 6;
        float v = 0.f;
        if (n < 18)      v = w_p[n * KDIM + ci * 9 + t];
        else if (n < 27) v = w_m[(n - 18) * KDIM + ci * 9 + t];
        Bpm[idx] = (_Float16)v;
    }
}

// Load one A fragment (v16h) from the LDS A-tile for K-step ks.
// A layout (contiguous, matches B packing): lane<16 -> K {0..15}; lane>=16 -> K {16..31}.
__device__ __forceinline__ v16h load_afrag(const _Float16* atile, int mrow,
                                           int khalf, int ks) {
    const v8h* p0 = (const v8h*)&atile[mrow * LDSROW + ks * 32 + khalf * 16];
    const v8h* p1 = (const v8h*)&atile[mrow * LDSROW + ks * 32 + khalf * 16 + 8];
    v8h lo = *p0, hi = *p1;
    return __builtin_shufflevector(lo, hi, 0, 1, 2, 3, 4, 5, 6, 7,
                                   8, 9, 10, 11, 12, 13, 14, 15);
}

// ---------------------------------------------------------------------------
// Kernel 1: fused offset(18ch) + mask(9ch) 3x3 conv via WMMA.
// One block = 16 pixels of one row (b, h, w0..w0+15); M=16, N=27(pad 32), K=576.
// ---------------------------------------------------------------------------
__global__ __launch_bounds__(128) void offmask_kernel(
    const float* __restrict__ x, const _Float16* __restrict__ Bpm,
    const float* __restrict__ b_p, const float* __restrict__ b_m,
    float* __restrict__ off, float* __restrict__ msk) {
    __shared__ _Float16 atile[16 * LDSROW];
    int bid = blockIdx.x;                 // 16*64*4 = 4096 blocks
    int b = bid >> 8;
    int rem = bid & 255;
    int h = rem >> 2;
    int w0 = (rem & 3) << 4;
    int tid = threadIdx.x;
    const float* xb = x + b * CIN * (HH * WW);

    // im2col staging: a[m][t*64+ci] = x[b][ci][h-1+ki][w0+m-1+kj], zero-padded.
    for (int idx = tid; idx < 16 * KTAP * 8; idx += 128) {  // 1152 tasks
        int mt = idx >> 3;
        int cb = (idx & 7) << 3;
        int m = mt / 9, t = mt - m * 9;
        int ki = t / 3, kj = t - ki * 3;
        int iy = h - 1 + ki;
        int ix = w0 + m - 1 + kj;
        bool valid = (iy >= 0) & (iy < HH) & (ix >= 0) & (ix < WW);
        int o = iy * WW + ix;
        v8h pack;
        for (int c = 0; c < 8; ++c) {
            float v = valid ? xb[(cb + c) * (HH * WW) + o] : 0.f;
            pack[c] = (_Float16)v;
        }
        *(v8h*)&atile[m * LDSROW + t * 64 + cb] = pack;
    }
    __syncthreads();

    int wave = tid >> 5, lane = tid & 31;
    if (wave < 2) {
        int nt = wave;
        int mrow = lane & 15, khalf = lane >> 4;
        v8f acc = {};
        for (int ks = 0; ks < KSTEPS; ++ks) {
            v16h a = load_afrag(atile, mrow, khalf, ks);
            v16h bf = *(const v16h*)&Bpm[((nt * KSTEPS + ks) * 32 + lane) * 16];
            acc = __builtin_amdgcn_wmma_f32_16x16x32_f16(
                false, a, false, bf, (short)0, acc, false, false);
        }
        int col = lane & 15;
        int mbase = khalf * 8;
        for (int r = 0; r < 8; ++r) {
            int M = r + mbase;
            int n = nt * 16 + col;
            int w = w0 + M;
            float v = acc[r];
            if (n < 18) {
                off[(b * 18 + n) * (HH * WW) + h * WW + w] = v + b_p[n];
            } else if (n < 27) {
                float s = v + b_m[n - 18];
                msk[(b * 9 + (n - 18)) * (HH * WW) + h * WW + w] =
                    1.f / (1.f + expf(-s));
            }
        }
    }
}

// ---------------------------------------------------------------------------
// Kernel 2: deformable conv. Bilinear-sample+modulate into LDS A-tile (f16),
// then 576->128 GEMM on WMMA. 4 waves x 2 channel-tiles each.
// ---------------------------------------------------------------------------
__global__ __launch_bounds__(128) void deform_kernel(
    const float* __restrict__ x, const float* __restrict__ off,
    const float* __restrict__ msk, const _Float16* __restrict__ Bd,
    const float* __restrict__ b_d, float* __restrict__ out) {
    __shared__ _Float16 atile[16 * LDSROW];
    int bid = blockIdx.x;
    int b = bid >> 8;
    int rem = bid & 255;
    int h = rem >> 2;
    int w0 = (rem & 3) << 4;
    int tid = threadIdx.x;
    const float* xb = x + b * CIN * (HH * WW);

    for (int idx = tid; idx < 16 * KTAP * 8; idx += 128) {
        int mt = idx >> 3;
        int cb = (idx & 7) << 3;
        int m = mt / 9, t = mt - m * 9;
        int ki = t / 3, kj = t - ki * 3;
        int w = w0 + m;
        int pix = h * WW + w;
        float oy = off[(b * 18 + 2 * t) * (HH * WW) + pix];
        float ox = off[(b * 18 + 2 * t + 1) * (HH * WW) + pix];
        float mv = msk[(b * 9 + t) * (HH * WW) + pix];
        float py = (float)(h - 1 + ki) + oy;
        float px = (float)(w - 1 + kj) + ox;
        float fy = floorf(py), fx = floorf(px);
        float ly = py - fy, lx = px - fx;
        int y0 = (int)fy, x0 = (int)fx;
        int y1 = y0 + 1, x1 = x0 + 1;
        // zero weights for out-of-range corners (torchvision v2 semantics)
        float vy0 = (y0 >= 0 && y0 < HH) ? 1.f : 0.f;
        float vy1 = (y1 >= 0 && y1 < HH) ? 1.f : 0.f;
        float vx0 = (x0 >= 0 && x0 < WW) ? 1.f : 0.f;
        float vx1 = (x1 >= 0 && x1 < WW) ? 1.f : 0.f;
        float w00 = (1.f - ly) * (1.f - lx) * vy0 * vx0;
        float w01 = (1.f - ly) * lx * vy0 * vx1;
        float w10 = ly * (1.f - lx) * vy1 * vx0;
        float w11 = ly * lx * vy1 * vx1;
        int y0c = min(max(y0, 0), HH - 1), y1c = min(max(y1, 0), HH - 1);
        int x0c = min(max(x0, 0), WW - 1), x1c = min(max(x1, 0), WW - 1);
        int o00 = y0c * WW + x0c, o01 = y0c * WW + x1c;
        int o10 = y1c * WW + x0c, o11 = y1c * WW + x1c;
        v8h pack;
        for (int c = 0; c < 8; ++c) {
            const float* xc = xb + (cb + c) * (HH * WW);
            float v = w00 * xc[o00] + w01 * xc[o01] +
                      w10 * xc[o10] + w11 * xc[o11];
            pack[c] = (_Float16)(v * mv);
        }
        *(v8h*)&atile[m * LDSROW + t * 64 + cb] = pack;
    }
    __syncthreads();

    int wave = tid >> 5, lane = tid & 31;
    int mrow = lane & 15, khalf = lane >> 4;
    int nt0 = wave * 2, nt1 = wave * 2 + 1;
    v8f acc0 = {}, acc1 = {};
    for (int ks = 0; ks < KSTEPS; ++ks) {
        v16h a = load_afrag(atile, mrow, khalf, ks);
        v16h bf0 = *(const v16h*)&Bd[((nt0 * KSTEPS + ks) * 32 + lane) * 16];
        acc0 = __builtin_amdgcn_wmma_f32_16x16x32_f16(
            false, a, false, bf0, (short)0, acc0, false, false);
        v16h bf1 = *(const v16h*)&Bd[((nt1 * KSTEPS + ks) * 32 + lane) * 16];
        acc1 = __builtin_amdgcn_wmma_f32_16x16x32_f16(
            false, a, false, bf1, (short)0, acc1, false, false);
    }
    int col = lane & 15;
    int mbase = khalf * 8;
    for (int r = 0; r < 8; ++r) {
        int M = r + mbase;
        int w = w0 + M;
        int n0 = nt0 * 16 + col;
        int n1 = nt1 * 16 + col;
        out[(b * COUT + n0) * (HH * WW) + h * WW + w] = acc0[r] + b_d[n0];
        out[(b * COUT + n1) * (HH * WW) + h * WW + w] = acc1[r] + b_d[n1];
    }
}

// ---------------------------------------------------------------------------
// Kernel 3: per-channel batch stats (one block per channel, tree reduce).
// stats[co] = mean, stats[128+co] = rsqrt(var + eps)
// ---------------------------------------------------------------------------
__global__ __launch_bounds__(256) void bn_stats_kernel(
    const float* __restrict__ y, float* __restrict__ stats) {
    __shared__ float ssum[256];
    __shared__ float ssq[256];
    int co = blockIdx.x;
    int tid = threadIdx.x;
    float s = 0.f, q = 0.f;
    for (int idx = tid; idx < BB * HH * WW; idx += 256) {
        int bb = idx >> 12;
        int hw = idx & 4095;
        float v = y[(bb * COUT + co) * (HH * WW) + hw];
        s += v;
        q += v * v;
    }
    ssum[tid] = s;
    ssq[tid] = q;
    __syncthreads();
    for (int st = 128; st > 0; st >>= 1) {
        if (tid < st) {
            ssum[tid] += ssum[tid + st];
            ssq[tid] += ssq[tid + st];
        }
        __syncthreads();
    }
    if (tid == 0) {
        const float inv_n = 1.f / (float)(BB * HH * WW);
        float mean = ssum[0] * inv_n;
        float var = ssq[0] * inv_n - mean * mean;
        stats[co] = mean;
        stats[COUT + co] = rsqrtf(var + 1e-5f);
    }
}

// ---------------------------------------------------------------------------
// Kernel 4: normalize + affine + exact-erf GELU, in place.
// ---------------------------------------------------------------------------
__global__ void bn_gelu_kernel(float* __restrict__ y,
                               const float* __restrict__ stats,
                               const float* __restrict__ gamma,
                               const float* __restrict__ beta) {
    int i = blockIdx.x * blockDim.x + threadIdx.x;
    const int N = BB * COUT * HH * WW;
    if (i >= N) return;
    int co = (i >> 12) & (COUT - 1);
    float v = y[i];
    float yn = (v - stats[co]) * stats[COUT + co] * gamma[co] + beta[co];
    y[i] = 0.5f * yn * (1.f + erff(yn * 0.70710678118654752f));
}

extern "C" void kernel_launch(void* const* d_in, const int* in_sizes, int n_in,
                              void* d_out, int out_size, void* d_ws,
                              size_t ws_size, hipStream_t stream) {
    (void)in_sizes; (void)n_in; (void)out_size; (void)ws_size;
    const float* x     = (const float*)d_in[0];
    const float* w_p   = (const float*)d_in[1];
    const float* b_p   = (const float*)d_in[2];
    const float* w_m   = (const float*)d_in[3];
    const float* b_m   = (const float*)d_in[4];
    const float* w_d   = (const float*)d_in[5];
    const float* b_d   = (const float*)d_in[6];
    const float* gamma = (const float*)d_in[7];
    const float* beta  = (const float*)d_in[8];
    float* y = (float*)d_out;

    // Workspace layout (bytes)
    char* ws = (char*)d_ws;
    float*     off   = (float*)ws;                    // 16*18*4096 f32 = 4,718,592 B
    float*     msk   = (float*)(ws + 4718592);        // 16*9*4096  f32 = 2,359,296 B
    _Float16*  Bpm   = (_Float16*)(ws + 7077888);     // 18,432 f16 = 36,864 B
    _Float16*  Bd    = (_Float16*)(ws + 7114752);     // 73,728 f16 = 147,456 B
    float*     stats = (float*)(ws + 7262208);        // 256 f32

    pack_weights_kernel<<<64, 256, 0, stream>>>(w_p, w_m, w_d, Bpm, Bd);
    offmask_kernel<<<4096, 128, 0, stream>>>(x, Bpm, b_p, b_m, off, msk);
    deform_kernel<<<4096, 128, 0, stream>>>(x, off, msk, Bd, b_d, y);
    bn_stats_kernel<<<COUT, 256, 0, stream>>>(y, stats);
    const int N = BB * COUT * HH * WW;
    bn_gelu_kernel<<<(N + 255) / 256, 256, 0, stream>>>(y, stats, gamma, beta);
}
